// SelfAttention2d_18537078850120
// MI455X (gfx1250) — matrix operand order repaired and checked by reading, not compiled, not run on hardware
//
#include <hip/hip_runtime.h>
#include <hip/hip_bf16.h>

// ---------------------------------------------------------------------------
// SelfAttention2d on MI455X (gfx1250): GroupNorm -> QKV -> flash-attention
// (v_wmma_f32_16x16x32_f16 + global_load_async_to_lds_b128 K/V staging)
// -> out-proj -> residual.  n=4, c=256, h=w=64 -> hw=4096, n_head=4, dh=64.
// ---------------------------------------------------------------------------

#define N_S   4
#define C_CH  256
#define HW    4096
#define NH    4
#define DH    64
#define CHW   (C_CH * HW)      // 1048576 per sample

typedef _Float16 h8  __attribute__((ext_vector_type(8)));
typedef _Float16 h16 __attribute__((ext_vector_type(16)));
typedef float    f4  __attribute__((ext_vector_type(4)));
typedef float    f8  __attribute__((ext_vector_type(8)));

#define CMB16(a,b) __builtin_shufflevector(a,b,0,1,2,3,4,5,6,7,8,9,10,11,12,13,14,15)
#define WMMA_F16(a,b,c) __builtin_amdgcn_wmma_f32_16x16x32_f16(false,(a),false,(b),(short)0,(c),false,false)

// A-matrix (16x32, row-major memory [M][K]): lane m=l&15 is row M,
// elements 0-7 = K(8*hi..), 8-15 = K(16+8*hi..)
__device__ __forceinline__ h16 load_mat_a(const _Float16* p, int stride, int m, int hi) {
  const _Float16* r = p + m * stride + 8 * hi;
  h8 x0 = *(const h8*)(r);
  h8 x1 = *(const h8*)(r + 16);
  return CMB16(x0, x1);
}
// B-matrix (32x16, column-major memory [N][K]): lane m=l&15 is column N,
// elements 0-7 = K(8*hi..), 8-15 = K(16+8*hi..)
__device__ __forceinline__ h16 load_mat_b(const _Float16* p, int stride, int m, int hi) {
  const _Float16* r = p + m * stride + 8 * hi;
  h8 x0 = *(const h8*)(r);
  h8 x1 = *(const h8*)(r + 16);
  return CMB16(x0, x1);
}

__device__ __forceinline__ float red_max16(float v) {
  v = fmaxf(v, __shfl_xor(v, 1));
  v = fmaxf(v, __shfl_xor(v, 2));
  v = fmaxf(v, __shfl_xor(v, 4));
  v = fmaxf(v, __shfl_xor(v, 8));
  return v;
}

// Async global->LDS copy (CDNA5 TDM-lite path, ASYNCcnt-tracked).
// LDS aperture: flat addr[31:0] IS the LDS byte offset (ISA 10.2), so the
// VDST lds-address VGPR is just the truncated generic pointer.
__device__ __forceinline__ void async_ld_b128(const _Float16* gptr, const _Float16* lptr) {
  unsigned int loff = (unsigned int)(unsigned long long)lptr;
  asm volatile("global_load_async_to_lds_b128 %0, %1, off"
               :: "v"(loff), "v"((unsigned long long)gptr) : "memory");
}
#define WAIT_ASYNC_2() asm volatile("s_wait_asynccnt 0x2" ::: "memory")
#define WAIT_ASYNC_0() asm volatile("s_wait_asynccnt 0x0" ::: "memory")

// ---------------------------------------------------------------------------
// 1) GroupNorm stats: per-sample partial sums (deterministic, no float atomics)
// ---------------------------------------------------------------------------
__global__ __launch_bounds__(256) void sa2d_stats_partial(const float* __restrict__ x,
                                                          float* __restrict__ part) {
  int n = blockIdx.x;          // sample
  int blk = blockIdx.y;        // 0..31
  const float* xb = x + (size_t)n * CHW + (size_t)blk * (CHW / 32);
  float s = 0.f, s2 = 0.f;
  for (int i = threadIdx.x; i < CHW / 32; i += 256) {
    float v = xb[i];
    s += v;
    s2 += v * v;
  }
  for (int m = 1; m < 32; m <<= 1) {
    s  += __shfl_xor(s, m);
    s2 += __shfl_xor(s2, m);
  }
  __shared__ float sd[16];
  int wave = threadIdx.x >> 5;
  if ((threadIdx.x & 31) == 0) { sd[wave * 2] = s; sd[wave * 2 + 1] = s2; }
  __syncthreads();
  if (threadIdx.x == 0) {
    float ts = 0.f, ts2 = 0.f;
    for (int w = 0; w < 8; ++w) { ts += sd[w * 2]; ts2 += sd[w * 2 + 1]; }
    part[(n * 32 + blk) * 2]     = ts;
    part[(n * 32 + blk) * 2 + 1] = ts2;
  }
}

__global__ void sa2d_stats_final(const float* __restrict__ part, float* __restrict__ stats) {
  int n = blockIdx.x;
  int l = threadIdx.x;         // 32 threads
  float s  = part[(n * 32 + l) * 2];
  float s2 = part[(n * 32 + l) * 2 + 1];
  for (int m = 1; m < 32; m <<= 1) {
    s  += __shfl_xor(s, m);
    s2 += __shfl_xor(s2, m);
  }
  if (l == 0) {
    float mean = s / (float)CHW;
    float var  = s2 / (float)CHW - mean * mean;
    stats[n * 2]     = mean;
    stats[n * 2 + 1] = rsqrtf(var + 1e-5f);
  }
}

// ---------------------------------------------------------------------------
// 2) Weight f32 -> f16 cast
// ---------------------------------------------------------------------------
__global__ __launch_bounds__(256) void sa2d_wcast(const float* __restrict__ wqkv,
                                                  const float* __restrict__ wout,
                                                  _Float16* __restrict__ wqkv16,
                                                  _Float16* __restrict__ wout16) {
  int i = blockIdx.x * 256 + threadIdx.x;
  if (i < 3 * C_CH * C_CH) wqkv16[i] = (_Float16)wqkv[i];
  if (i < C_CH * C_CH)     wout16[i] = (_Float16)wout[i];
}

// ---------------------------------------------------------------------------
// 3) Normalize + affine, transpose to pixel-major f16: xn[n][hw][c]
// ---------------------------------------------------------------------------
__global__ __launch_bounds__(256) void sa2d_norm(const float* __restrict__ x,
                                                 const float* __restrict__ gamma,
                                                 const float* __restrict__ beta,
                                                 const float* __restrict__ stats,
                                                 _Float16* __restrict__ xn) {
  int n = blockIdx.y;
  int p = blockIdx.x * 256 + threadIdx.x;   // grid.x = 16 -> 4096 pixels
  float mean = stats[n * 2];
  float rstd = stats[n * 2 + 1];
  const float* xb = x + (size_t)n * CHW + p;
  _Float16* yb = xn + ((size_t)n * HW + p) * C_CH;
#pragma unroll 4
  for (int c0 = 0; c0 < C_CH; c0 += 8) {
    h8 v;
#pragma unroll
    for (int cc = 0; cc < 8; ++cc) {
      int c = c0 + cc;
      float t = (xb[(size_t)c * HW] - mean) * rstd * gamma[c] + beta[c];
      v[cc] = (_Float16)t;
    }
    *(h8*)(yb + c0) = v;
  }
}

// ---------------------------------------------------------------------------
// 4) QKV GEMM (WMMA): D[p,o] = Xn[p,:] . Wqkv[o,:]  (M=pixel, N=out-chan, K=256)
//    Q,K -> [n][nh][hw][dh] (scale folded into Q), V -> [n][nh][dh][hw]
// ---------------------------------------------------------------------------
__global__ __launch_bounds__(256) void sa2d_qkv(const _Float16* __restrict__ xn,
                                                const _Float16* __restrict__ w16,
                                                const float* __restrict__ bq,
                                                _Float16* __restrict__ Q,
                                                _Float16* __restrict__ K,
                                                _Float16* __restrict__ Vt) {
  int wave = threadIdx.x >> 5;
  int lane = threadIdx.x & 31;
  int hi = lane >> 4, m = lane & 15;
  int n = blockIdx.z;
  int p0 = blockIdx.x * 32 + (wave & 1) * 16;   // grid.x = 128
  int o0 = blockIdx.y * 64 + (wave >> 1) * 16;  // grid.y = 12
  const _Float16* A = xn + ((size_t)n * HW + p0) * C_CH;
  const _Float16* B = w16 + (size_t)o0 * C_CH;
  f8 c = {};
#pragma unroll
  for (int k = 0; k < C_CH; k += 32) {
    h16 a = load_mat_a(A + k, C_CH, m, hi);
    h16 b = load_mat_b(B + k, C_CH, m, hi);
    c = WMMA_F16(a, b, c);
  }
  int o = o0 + m;
  float bias = bq[o];
  float vals[8];
#pragma unroll
  for (int r = 0; r < 8; ++r) vals[r] = c[r] + bias;

  if (o < C_CH) {                        // Q, fold dh^-0.5 = 0.125
    int head = o >> 6, dh = o & 63;
    _Float16* qb = Q + ((size_t)(n * NH + head)) * HW * DH + dh;
#pragma unroll
    for (int r = 0; r < 8; ++r) qb[(size_t)(p0 + r + 8 * hi) * DH] = (_Float16)(vals[r] * 0.125f);
  } else if (o < 2 * C_CH) {             // K
    int oo = o - C_CH; int head = oo >> 6, dh = oo & 63;
    _Float16* kb = K + ((size_t)(n * NH + head)) * HW * DH + dh;
#pragma unroll
    for (int r = 0; r < 8; ++r) kb[(size_t)(p0 + r + 8 * hi) * DH] = (_Float16)vals[r];
  } else {                               // V transposed [dh][hw]: packed b128 store
    int oo = o - 2 * C_CH; int head = oo >> 6, dh = oo & 63;
    h8 vv;
#pragma unroll
    for (int r = 0; r < 8; ++r) vv[r] = (_Float16)vals[r];
    *(h8*)(Vt + (((size_t)(n * NH + head)) * DH + dh) * HW + p0 + 8 * hi) = vv;
  }
}

// ---------------------------------------------------------------------------
// 5) Flash attention. Workgroup = 8 waves x 16 queries = 128 queries.
//    K/V blocks (32 keys) staged into LDS with async global->LDS copies,
//    double-buffered. Per 32 keys: 4 WMMA (S) + 1 WMMA (row-sum via ones
//    B-tile) + 4 WMMA (P.V). Only the row-max uses lane shuffles.
// ---------------------------------------------------------------------------
__global__ __launch_bounds__(256) void sa2d_attn(const _Float16* __restrict__ Q,
                                                 const _Float16* __restrict__ K,
                                                 const _Float16* __restrict__ Vt,
                                                 _Float16* __restrict__ Y) {
  __shared__ _Float16 kbuf[2][32 * DH];   // [key][dh]   4 KB x2
  __shared__ _Float16 vbuf[2][DH * 32];   // [dh][key]   4 KB x2
  __shared__ _Float16 pbuf[8][16 * 32];   // per-wave P tile, 1 KB each
  int tid = threadIdx.x;
  int wave = tid >> 5;
  int lane = tid & 31;
  int hi = lane >> 4, m = lane & 15;
  int headIdx = blockIdx.x;               // n*NH + h, 0..15
  int nIdx = headIdx >> 2, h = headIdx & 3;
  int q0 = blockIdx.y * 128 + wave * 16;  // grid.y = 32
  const _Float16* Qb = Q  + (size_t)headIdx * HW * DH;
  const _Float16* Kb = K  + (size_t)headIdx * HW * DH;
  const _Float16* Vb = Vt + (size_t)headIdx * DH * HW;
  _Float16* lw = pbuf[wave];

  // cooperative-copy coordinates (256 threads, 16B each => 4KB per issue)
  int kRow = tid >> 3, kOff = (tid & 7) * 8;   // K block: 32 rows x 64 halves
  int vRow = tid >> 2, vOff = (tid & 3) * 8;   // V block: 64 rows x 32 halves

  h16 qa0 = load_mat_a(Qb + (size_t)q0 * DH,      DH, m, hi);
  h16 qa1 = load_mat_a(Qb + (size_t)q0 * DH + 32, DH, m, hi);

  h16 ones;
#pragma unroll
  for (int i = 0; i < 16; ++i) ones[i] = (_Float16)1.0f;

  f8 acc0 = {}, acc1 = {}, acc2 = {}, acc3 = {}, accL = {};
  float mrow[8];
#pragma unroll
  for (int r = 0; r < 8; ++r) mrow[r] = -3.0e38f;

  // prefetch first K/V block
  async_ld_b128(Kb + (size_t)kRow * DH + kOff, &kbuf[0][kRow * DH + kOff]);
  async_ld_b128(Vb + (size_t)vRow * HW + vOff, &vbuf[0][vRow * 32 + vOff]);

  for (int j = 0, it = 0; j < HW; j += 32, ++it) {
    int cur = it & 1;
    if (j + 32 < HW) {   // issue next block before waiting on current
      int jn = j + 32;
      async_ld_b128(Kb + (size_t)(jn + kRow) * DH + kOff, &kbuf[cur ^ 1][kRow * DH + kOff]);
      async_ld_b128(Vb + (size_t)vRow * HW + jn + vOff,   &vbuf[cur ^ 1][vRow * 32 + vOff]);
      WAIT_ASYNC_2();    // current block's 2 copies complete (in-order)
    } else {
      WAIT_ASYNC_0();
    }
    __syncthreads();     // all waves' copies for `cur` are in LDS

    const _Float16* kb = kbuf[cur];
    const _Float16* vb = vbuf[cur];
    h16 kb00 = load_mat_b(kb + 0,         DH, m, hi);   // keys 0-15,  dh 0-31
    h16 kb01 = load_mat_b(kb + 32,        DH, m, hi);   // keys 0-15,  dh 32-63
    h16 kb10 = load_mat_b(kb + 1024,      DH, m, hi);   // keys 16-31, dh 0-31
    h16 kb11 = load_mat_b(kb + 1024 + 32, DH, m, hi);   // keys 16-31, dh 32-63
    f8 s0 = {}, s1 = {};
    s0 = WMMA_F16(qa0, kb00, s0);
    s0 = WMMA_F16(qa1, kb01, s0);
    s1 = WMMA_F16(qa0, kb10, s1);
    s1 = WMMA_F16(qa1, kb11, s1);

#pragma unroll
    for (int r = 0; r < 8; ++r) {
      float mx = red_max16(fmaxf(s0[r], s1[r]));
      float mnew = fmaxf(mrow[r], mx);
      float corr = __expf(mrow[r] - mnew);
      float p0 = __expf(s0[r] - mnew);
      float p1 = __expf(s1[r] - mnew);
      mrow[r] = mnew;
      acc0[r] *= corr; acc1[r] *= corr; acc2[r] *= corr; acc3[r] *= corr;
      accL[r] *= corr;
      // C-layout -> LDS scalar layout: row M = r + 8*hi, col N = m (+16 for s1)
      lw[(r + 8 * hi) * 32 + m]      = (_Float16)p0;
      lw[(r + 8 * hi) * 32 + 16 + m] = (_Float16)p1;
    }
    // Re-read P in A-layout (DS ops are in-order within a wave)
    h8 x0 = *(const h8*)(lw + m * 32 + 8 * hi);
    h8 x1 = *(const h8*)(lw + m * 32 + 16 + 8 * hi);
    h16 pa = CMB16(x0, x1);
    h16 vb0 = load_mat_b(vb + 0 * 512, 32, m, hi);
    h16 vb1 = load_mat_b(vb + 1 * 512, 32, m, hi);
    h16 vb2 = load_mat_b(vb + 2 * 512, 32, m, hi);
    h16 vb3 = load_mat_b(vb + 3 * 512, 32, m, hi);
    accL = WMMA_F16(pa, ones, accL);    // row-sum of P on the matrix pipe
    acc0 = WMMA_F16(pa, vb0, acc0);
    acc1 = WMMA_F16(pa, vb1, acc1);
    acc2 = WMMA_F16(pa, vb2, acc2);
    acc3 = WMMA_F16(pa, vb3, acc3);

    __syncthreads();     // everyone done reading `cur` before it is re-filled
  }

#pragma unroll
  for (int r = 0; r < 8; ++r) {
    float inv = 1.0f / accL[r];
    int p = q0 + r + 8 * hi;
    _Float16* yp = Y + ((size_t)(nIdx * HW + p)) * C_CH + h * DH + m;
    yp[0]  = (_Float16)(acc0[r] * inv);
    yp[16] = (_Float16)(acc1[r] * inv);
    yp[32] = (_Float16)(acc2[r] * inv);
    yp[48] = (_Float16)(acc3[r] * inv);
  }
}

// ---------------------------------------------------------------------------
// 6) Output projection GEMM (WMMA) + bias + residual -> f32 out [n][c][hw]
// ---------------------------------------------------------------------------
__global__ __launch_bounds__(256) void sa2d_proj(const _Float16* __restrict__ y16,
                                                 const _Float16* __restrict__ w16,
                                                 const float* __restrict__ bo,
                                                 const float* __restrict__ x,
                                                 float* __restrict__ out) {
  int wave = threadIdx.x >> 5;
  int lane = threadIdx.x & 31;
  int hi = lane >> 4, m = lane & 15;
  int n = blockIdx.z;
  int p0 = blockIdx.x * 32 + (wave & 1) * 16;   // grid.x = 128
  int o0 = blockIdx.y * 64 + (wave >> 1) * 16;  // grid.y = 4
  const _Float16* A = y16 + ((size_t)n * HW + p0) * C_CH;
  const _Float16* B = w16 + (size_t)o0 * C_CH;
  f8 c = {};
#pragma unroll
  for (int k = 0; k < C_CH; k += 32) {
    h16 a = load_mat_a(A + k, C_CH, m, hi);
    h16 b = load_mat_b(B + k, C_CH, m, hi);
    c = WMMA_F16(a, b, c);
  }
  int o = o0 + m;
  float bias = bo[o];
  const float* xb = x + ((size_t)(n * C_CH + o)) * HW;
  float* ob = out + ((size_t)(n * C_CH + o)) * HW;
  int p = p0 + 8 * hi;                  // 8 consecutive pixels per lane
  f4 r0 = *(const f4*)(xb + p);
  f4 r1 = *(const f4*)(xb + p + 4);
  f4 w0, w1;
#pragma unroll
  for (int i = 0; i < 4; ++i) { w0[i] = c[i] + bias + r0[i]; w1[i] = c[4 + i] + bias + r1[i]; }
  *(f4*)(ob + p)     = w0;
  *(f4*)(ob + p + 4) = w1;
}

// ---------------------------------------------------------------------------
extern "C" void kernel_launch(void* const* d_in, const int* in_sizes, int n_in,
                              void* d_out, int out_size, void* d_ws, size_t ws_size,
                              hipStream_t stream) {
  const float* x     = (const float*)d_in[0];
  const float* gamma = (const float*)d_in[1];
  const float* beta  = (const float*)d_in[2];
  const float* wqkv  = (const float*)d_in[3];
  const float* bqkv  = (const float*)d_in[4];
  const float* wout  = (const float*)d_in[5];
  const float* bout  = (const float*)d_in[6];
  float* out = (float*)d_out;

  char* ws = (char*)d_ws;
  float*    stats = (float*)(ws);                       // 8 floats
  float*    part  = (float*)(ws + 1024);                // 256 floats
  _Float16* wq16  = (_Float16*)(ws + 4096);             // 768*256 halves
  _Float16* wo16  = (_Float16*)(ws + 4096 + 3 * C_CH * C_CH * 2);
  const size_t MB = 1u << 20;
  _Float16* xn16  = (_Float16*)(ws + 1 * MB);           // [n][hw][c]  8 MB (reused as y16)
  _Float16* q16   = (_Float16*)(ws + 9 * MB);           // [n][nh][hw][dh] 8 MB
  _Float16* k16   = (_Float16*)(ws + 17 * MB);          // [n][nh][hw][dh] 8 MB
  _Float16* v16   = (_Float16*)(ws + 25 * MB);          // [n][nh][dh][hw] 8 MB

  sa2d_stats_partial<<<dim3(N_S, 32), 256, 0, stream>>>(x, part);
  sa2d_stats_final<<<N_S, 32, 0, stream>>>(part, stats);
  sa2d_wcast<<<(3 * C_CH * C_CH + 255) / 256, 256, 0, stream>>>(wqkv, wout, wq16, wo16);
  sa2d_norm<<<dim3(HW / 256, N_S), 256, 0, stream>>>(x, gamma, beta, stats, xn16);
  sa2d_qkv<<<dim3(HW / 32, (3 * C_CH) / 64, N_S), 256, 0, stream>>>(xn16, wq16, bqkv,
                                                                    q16, k16, v16);
  sa2d_attn<<<dim3(N_S * NH, HW / 128), 256, 0, stream>>>(q16, k16, v16, xn16 /* y16 */);
  sa2d_proj<<<dim3(HW / 32, C_CH / 64, N_S), 256, 0, stream>>>(xn16 /* y16 */, wo16, bout,
                                                               x, out);
}
